// MoeLayer_35596688949259
// MI455X (gfx1250) — matrix operand order repaired and checked by reading, not compiled, not run on hardware
//
#include <hip/hip_runtime.h>
#include <hip/hip_bf16.h>

// ---- problem constants (match reference) ----
#define NB 2
#define NS 2048
#define NTOK (NB * NS)        // 4096 tokens
#define DIN 1024
#define DOUT 1024
#define NEXP 8
#define TOPK 2
#define CAP NTOK              // a token appears at most once per expert list

typedef __attribute__((ext_vector_type(16))) _Float16 v16h;
typedef __attribute__((ext_vector_type(8)))  _Float16 v8h;
typedef __attribute__((ext_vector_type(8)))  float    v8f;

// -------------------------------------------------------------------------
// Kernel 1: fp32 -> f16 bulk conversion (n must be a multiple of 8)
// -------------------------------------------------------------------------
__global__ __launch_bounds__(256) void moe_cvt_f16(const float* __restrict__ s,
                                                   _Float16* __restrict__ d,
                                                   int n) {
    int i = (blockIdx.x * 256 + threadIdx.x) * 8;
    if (i >= n) return;
    float4 f0 = *(const float4*)(s + i);
    float4 f1 = *(const float4*)(s + i + 4);
    v8h h;
    h[0] = (_Float16)f0.x; h[1] = (_Float16)f0.y;
    h[2] = (_Float16)f0.z; h[3] = (_Float16)f0.w;
    h[4] = (_Float16)f1.x; h[5] = (_Float16)f1.y;
    h[6] = (_Float16)f1.z; h[7] = (_Float16)f1.w;
    *(v8h*)(d + i) = h;   // 16-byte store
}

// -------------------------------------------------------------------------
// Kernel 2: gate logits + top-2 + softmax + scatter into per-expert lists.
// One wave32 per token; 8 waves (256 threads) per block. Full fp32 gating.
// -------------------------------------------------------------------------
__global__ __launch_bounds__(256) void moe_route(const float* __restrict__ x,
                                                 const float* __restrict__ Wg,
                                                 int* __restrict__ cnt,
                                                 int* __restrict__ tok,
                                                 float* __restrict__ wts) {
    __shared__ float sWg[NEXP * DIN];   // 32 KB
    int tid = threadIdx.x;
    for (int i = tid; i < NEXP * DIN; i += 256) sWg[i] = Wg[i];
    __syncthreads();

    int wave = tid >> 5, lane = tid & 31;
    int t = blockIdx.x * 8 + wave;
    const float* xr = x + (size_t)t * DIN;

    float acc[NEXP];
#pragma unroll
    for (int e = 0; e < NEXP; ++e) acc[e] = 0.f;

    for (int it = 0; it < DIN / 32; ++it) {
        int d = lane + it * 32;
        float xv = xr[d];                 // coalesced per wave
#pragma unroll
        for (int e = 0; e < NEXP; ++e)
            acc[e] = fmaf(xv, sWg[e * DIN + d], acc[e]);
    }
    // wave32 butterfly reduction -> every lane holds full dot products
#pragma unroll
    for (int e = 0; e < NEXP; ++e)
#pragma unroll
        for (int off = 16; off > 0; off >>= 1)
            acc[e] += __shfl_xor(acc[e], off, 32);

    if (lane == 0) {
        // top-2 with strict '>' scan: matches lax.top_k tie ordering
        float b0 = -3.4e38f, b1 = -3.4e38f;
        int i0 = 0, i1 = 0;
#pragma unroll
        for (int e = 0; e < NEXP; ++e) {
            float v = acc[e];
            if (v > b0)      { b1 = b0; i1 = i0; b0 = v; i0 = e; }
            else if (v > b1) { b1 = v; i1 = e; }
        }
        // softmax over the two selected logits (fp32)
        float r  = __expf(b1 - b0);
        float w0 = 1.0f / (1.0f + r);
        float w1 = r * w0;
        int s0 = atomicAdd(&cnt[i0], 1);
        tok[i0 * CAP + s0] = t;  wts[i0 * CAP + s0] = w0;
        int s1 = atomicAdd(&cnt[i1], 1);
        tok[i1 * CAP + s1] = t;  wts[i1 * CAP + s1] = w1;
    }
}

// -------------------------------------------------------------------------
// Kernel 3: gathered expert GEMM with v_wmma_f32_16x16x32_f16.
// grid = (DOUT/128, CAP/16, NEXP); block = 256 (8 waves).
// Each block: 16 gathered tokens x 128 output features for one expert.
// -------------------------------------------------------------------------
#define A_STRIDE 1032   // halves; 16B shift per row -> conflict-free b128 LDS reads

__global__ __launch_bounds__(256) void moe_gemm(const _Float16* __restrict__ xh,
                                                const _Float16* __restrict__ weh,
                                                const float* __restrict__ be,
                                                const int* __restrict__ cnt,
                                                const int* __restrict__ tok,
                                                const float* __restrict__ wts,
                                                float* __restrict__ out) {
    __shared__ _Float16 As[16 * A_STRIDE];   // ~33 KB A panel (16 x 1024 f16)
    __shared__ int   stok[16];
    __shared__ float sw[16];

    int e = blockIdx.z;
    int count = cnt[e];
    int m0 = blockIdx.y * 16;
    if (m0 >= count) return;                 // inactive tile -> cheap exit

    int tid = threadIdx.x;
    if (tid < 16) {
        int idx = m0 + tid;
        if (idx < count) { stok[tid] = tok[e * CAP + idx]; sw[tid] = wts[e * CAP + idx]; }
        else             { stok[tid] = 0;                  sw[tid] = 0.0f; }
    }
    __syncthreads();

    // Gather A panel into LDS: 16 rows x 1024 halves = 2048 x 16B chunks.
    for (int c = tid; c < 2048; c += 256) {
        int row = c >> 7;          // 128 chunks per row
        int cc  = c & 127;
        const uint4* src = (const uint4*)(xh + (size_t)stok[row] * DIN + cc * 8);
        *(uint4*)(&As[row * A_STRIDE + cc * 8]) = *src;
    }
    __syncthreads();

    int wave = tid >> 5, lane = tid & 31;
    int hi = lane >> 4;            // half-wave select (WMMA K split)
    int lm = lane & 15;
    int n  = blockIdx.x * 128 + wave * 16 + lm;   // output feature of this lane

    // B fragment source: We row n is contiguous in K (row-major [E,DOUT,DIN])
    const _Float16* wrow = weh + ((size_t)e * DOUT + n) * DIN + hi * 8;
    const _Float16* arow = &As[lm * A_STRIDE + hi * 8];

    v8f acc = {};
    for (int k = 0; k < DIN; k += 32) {
        // A frag (16x32 f16): lanes 0-15 row lm K[k..k+7],[k+16..k+23];
        //                     lanes 16-31 same row   K[k+8..k+15],[k+24..k+31]
        v8h a0 = *(const v8h*)(arow + k);        // ds_load_b128
        v8h a1 = *(const v8h*)(arow + k + 16);   // ds_load_b128
        // B frag (32x16 f16): lane's column n, same K split as A
        v8h b0 = *(const v8h*)(wrow + k);        // global_load_b128
        v8h b1 = *(const v8h*)(wrow + k + 16);   // global_load_b128
        v16h a = __builtin_shufflevector(a0, a1, 0,1,2,3,4,5,6,7,8,9,10,11,12,13,14,15);
        v16h b = __builtin_shufflevector(b0, b1, 0,1,2,3,4,5,6,7,8,9,10,11,12,13,14,15);
        acc = __builtin_amdgcn_wmma_f32_16x16x32_f16(
            false, a, false, b, (short)0, acc, false, false);
    }

    // C/D layout: VGPR r holds row (r + 8*hi), col lm. Scale by routing weight,
    // add expert bias, scatter-accumulate into the token's output row.
    float bias = be[e * DOUT + n];
    float* orow_base = out;
#pragma unroll
    for (int r = 0; r < 8; ++r) {
        int m = r + hi * 8;
        float w = sw[m];                       // 0 for padded tail rows
        float v = (acc[r] + bias) * w;
        float* p = orow_base + (size_t)stok[m] * DOUT + n;
        __hip_atomic_fetch_add(p, v, __ATOMIC_RELAXED, __HIP_MEMORY_SCOPE_AGENT);
    }
}

// -------------------------------------------------------------------------
// Launcher
// -------------------------------------------------------------------------
extern "C" void kernel_launch(void* const* d_in, const int* in_sizes, int n_in,
                              void* d_out, int out_size, void* d_ws, size_t ws_size,
                              hipStream_t stream) {
    const float* inputs = (const float*)d_in[0];   // [B,S,DIN] fp32
    const float* Wg     = (const float*)d_in[1];   // [E,DIN]   fp32
    const float* We     = (const float*)d_in[2];   // [E,DOUT,DIN] fp32
    const float* be     = (const float*)d_in[3];   // [E,DOUT]  fp32
    float* out = (float*)d_out;                    // [B,S,DOUT] fp32

    // workspace layout (~25.4 MB)
    char* ws = (char*)d_ws;
    const size_t XH_BYTES  = (size_t)NTOK * DIN * sizeof(_Float16);         // 8 MB
    const size_t WEH_BYTES = (size_t)NEXP * DOUT * DIN * sizeof(_Float16);  // 16 MB
    _Float16* xh  = (_Float16*)ws;
    _Float16* weh = (_Float16*)(ws + XH_BYTES);
    int*   cnt = (int*)(ws + XH_BYTES + WEH_BYTES);
    int*   tok = cnt + 64;                           // 256B pad after counters
    float* wts = (float*)(tok + NEXP * CAP);

    // zero output (atomic accumulation target) and expert counters
    hipMemsetAsync(out, 0, (size_t)out_size * sizeof(float), stream);
    hipMemsetAsync(cnt, 0, 64 * sizeof(int), stream);

    // 1) fp32 -> f16
    {
        int n1 = NTOK * DIN;
        moe_cvt_f16<<<n1 / (256 * 8), 256, 0, stream>>>(inputs, xh, n1);
        int n2 = NEXP * DOUT * DIN;
        moe_cvt_f16<<<n2 / (256 * 8), 256, 0, stream>>>(We, weh, n2);
    }
    // 2) routing (fp32 gate + top-2 softmax + scatter)
    moe_route<<<NTOK / 8, 256, 0, stream>>>(inputs, Wg, cnt, tok, wts);
    // 3) gathered expert GEMMs (WMMA f16 -> f32)
    dim3 grid(DOUT / 128, CAP / 16, NEXP);
    moe_gemm<<<grid, 256, 0, stream>>>(xh, weh, be, cnt, tok, wts, out);
}
